// PreconditionerNet_27797028340218
// MI455X (gfx1250) — matrix operand order repaired and checked by reading, not compiled, not run on hardware
//
#include <hip/hip_runtime.h>
#include <hip/hip_bf16.h>

typedef __attribute__((ext_vector_type(16))) _Float16 v16h;
typedef __attribute__((ext_vector_type(8)))  float    v8f;

// ---------------- layer 1: 1x1 conv 1->64 + PReLU (f32 in -> f16 out) ------
__global__ __launch_bounds__(256) void l1_conv1x1(
    const float* __restrict__ x, const float* __restrict__ w,
    const float* __restrict__ b, const float* __restrict__ alpha,
    _Float16* __restrict__ out, int total)
{
  int idx = blockIdx.x * 256 + threadIdx.x;
  if (idx >= total) return;
  int c   = idx & 63;
  int pos = idx >> 6;
  float v = x[pos] * w[c] + b[c];
  float a = *alpha;
  v = (v >= 0.f) ? v : a * v;
  out[idx] = (_Float16)v;
}

// ------------- weight f32 [K][COUT] -> f16 transposed [COUT][K] -------------
__global__ __launch_bounds__(256) void cvt_w_t(const float* __restrict__ s,
                                               _Float16* __restrict__ d,
                                               int K, int COUT)
{
  int idx = blockIdx.x * 256 + threadIdx.x;
  if (idx >= K * COUT) return;
  int k  = idx / COUT;
  int co = idx % COUT;
  d[(size_t)co * K + k] = (_Float16)s[idx];
}

// ---------------- 2x2 conv as implicit GEMM via WMMA ------------------------
// GEMM: M = B*Hout*Wout, N = COUT, K = 4*CIN (k = (ky*2+kx)*CIN + ci).
// Block: 256 thr = 8 waves; macro-tile 128(M) x 64(N).
// Wave w owns rows [16w,16w+16) x 64 cols -> 4 accumulators.
// Buffered K-step = 64 (two 32-wide WMMA chunks): 8 WMMAs per wave per
// barrier. LDS double-buffered. Activation (A) global loads are issued
// before the WMMA section (HBM latency overlapped with matrix work);
// weight (B) load+store are adjacent after it (L2-hot, no long live range).
template<int CIN, int COUT, int PADH, int PADW>
__global__ __launch_bounds__(256) void conv2x2_wmma(
    const _Float16* __restrict__ in,    // [B,Hin,Win,CIN]
    const _Float16* __restrict__ wgtT,  // [COUT][4*CIN] f16 (co-major)
    const float*    __restrict__ bias,  // [COUT]
    const float*    __restrict__ alpha, // scalar
    _Float16*       __restrict__ out,   // [B,Hout,Wout,COUT]
    int Bn, int Hin, int Win)
{
  const int K    = 4 * CIN;
  const int Hout = Hin + 2 * PADH - 1;
  const int Wout = Win + 2 * PADW - 1;
  const int M    = Bn * Hout * Wout;

  // row stride 72 halves = 144B = 9*16B -> every chunk stays 16B-aligned.
  __shared__ _Float16 Abuf[2][128][72];  // [m][k0..63]
  __shared__ _Float16 Bbuf[2][64][72];   // [n][k0..63] (pre-transposed wgt)

  const int tid    = threadIdx.x;
  const int m_base = blockIdx.x * 128;
  const int n_base = blockIdx.y * 64;

  // ---- A staging: 2 threads/row, 32 halves (4 x 16B) each ----
  const int arow = tid >> 1;             // 0..127
  const int asub = (tid & 1) << 5;       // 0 or 32 halves
  int am = m_base + arow; if (am >= M) am = M - 1;   // clamp; stores guarded
  const int awo = am % Wout;
  const int at  = am / Wout;
  const int aho = at % Hout;
  const int ab  = at / Hout;

  // ---- B staging: 4 threads/row, 16 halves (2 x 16B) each ----
  const int bnrow = tid >> 2;            // 0..63 (n)
  const int bksub = (tid & 3) << 4;      // 0,16,32,48 (k within step)
  const _Float16* wrow = wgtT + (size_t)(n_base + bnrow) * K + bksub;

  // ---- wave / fragment assignment (wave32 WMMA VGPR layouts) ----
  const int lane = tid & 31;
  const int wv   = tid >> 5;             // 0..7 -> M strip
  const int lhi  = lane >> 4;            // lane group 0/1
  const int llo  = lane & 15;
  const int mloc   = wv * 16 + llo;      // A-frag row
  const int kbaseA = lhi << 4;           // A: +16 k for lanes 16..31
  const int kbaseB = lhi << 4;           // B: +16 k for lanes 16..31

  v8f acc[4];
#pragma unroll
  for (int j = 0; j < 4; ++j) {
    float bv = bias[n_base + j * 16 + llo];
#pragma unroll
    for (int r = 0; r < 8; ++r) acc[j][r] = bv;
  }

  uint4 areg[4];                         // A staging registers (SW pipeline)

  auto stage_load_A = [&](int kk) {
    const int patch = kk / CIN;          // CIN % 64 == 0 -> fixed patch
    const int ci0   = kk % CIN;
    const int ky = patch >> 1, kx = patch & 1;
    const int hi = aho + ky - PADH;
    const int wi = awo + kx - PADW;
#pragma unroll
    for (int q = 0; q < 4; ++q) areg[q] = uint4{0u, 0u, 0u, 0u};
    if (hi >= 0 && hi < Hin && wi >= 0 && wi < Win) {
      const _Float16* src = in + (((size_t)ab * Hin + hi) * Win + wi) * CIN
                               + ci0 + asub;
#pragma unroll
      for (int q = 0; q < 4; ++q) areg[q] = *(const uint4*)(src + q * 8);
    }
  };
  auto stage_store_A = [&](int p) {
#pragma unroll
    for (int q = 0; q < 4; ++q)
      *(uint4*)&Abuf[p][arow][asub + q * 8] = areg[q];
  };
  auto copy_B = [&](int kk, int p) {     // load->store adjacent (L2-hot)
    uint4 b0 = *(const uint4*)(wrow + kk);
    uint4 b1 = *(const uint4*)(wrow + kk + 8);
    *(uint4*)&Bbuf[p][bnrow][bksub]     = b0;
    *(uint4*)&Bbuf[p][bnrow][bksub + 8] = b1;
  };

  stage_load_A(0);
  stage_store_A(0);
  copy_B(0, 0);
  __syncthreads();

  int p = 0;
#pragma unroll 1
  for (int kk = 0; kk < K; kk += 64) {
    const bool more = (kk + 64) < K;
    if (more) stage_load_A(kk + 64);     // HBM loads overlap WMMAs below

#pragma unroll
    for (int kc = 0; kc < 64; kc += 32) {
      v16h af;
#pragma unroll
      for (int h = 0; h < 8; ++h) {
        af[h]     = Abuf[p][mloc][kc + kbaseA + h];       // K 0..7 (+16 hi)
        af[h + 8] = Abuf[p][mloc][kc + kbaseA + 8 + h];   // K 8..15 (+16 hi)
      }
      v16h bf0, bf1, bf2, bf3;
#pragma unroll
      for (int h = 0; h < 16; ++h) {
        bf0[h] = Bbuf[p][llo][kc + kbaseB + h];
        bf1[h] = Bbuf[p][16 + llo][kc + kbaseB + h];
        bf2[h] = Bbuf[p][32 + llo][kc + kbaseB + h];
        bf3[h] = Bbuf[p][48 + llo][kc + kbaseB + h];
      }
      // Keep all fragment loads above, WMMAs below: one dscnt wait, then
      // four back-to-back WMMAs so the CDNA5 reuse-A (OPSEL RA) hint on
      // instructions 2..4 follows an identical immediately-preceding WMMA.
      __builtin_amdgcn_sched_barrier(0);
      acc[0] = __builtin_amdgcn_wmma_f32_16x16x32_f16(
          false, af, false, bf0, (short)0, acc[0], false, false);
      acc[1] = __builtin_amdgcn_wmma_f32_16x16x32_f16(
          false, af, false, bf1, (short)0, acc[1], true, false);
      acc[2] = __builtin_amdgcn_wmma_f32_16x16x32_f16(
          false, af, false, bf2, (short)0, acc[2], true, false);
      acc[3] = __builtin_amdgcn_wmma_f32_16x16x32_f16(
          false, af, false, bf3, (short)0, acc[3], true, false);
    }

    if (more) {
      stage_store_A(p ^ 1);              // fill the other buffer
      copy_B(kk + 64, p ^ 1);
      __syncthreads();                   // single barrier per 64-wide K-step
      p ^= 1;
    }
  }

  // ---- epilogue: PReLU, f16 store. D layout: VGPR r -> M = r + 8*lhi ----
  const float a = *alpha;
#pragma unroll
  for (int j = 0; j < 4; ++j) {
    const int n = n_base + j * 16 + llo;
#pragma unroll
    for (int r = 0; r < 8; ++r) {
      float v = acc[j][r];
      v = (v >= 0.f) ? v : a * v;
      const int m = m_base + wv * 16 + (lhi << 3) + r;
      if (m < M) {
        const int wo = m % Wout;
        const int t  = m / Wout;
        const int ho = t % Hout;
        const int bb = t / Hout;
        out[(((size_t)bb * Hout + ho) * Wout + wo) * COUT + n] = (_Float16)v;
      }
    }
  }
}

// ---------------- layer 6: 1x1 conv 64->1 + lower-triangle mask ------------
__global__ __launch_bounds__(256) void l6_conv1x1_mask(
    const _Float16* __restrict__ in,    // [B,256,256,64]
    const float* __restrict__ w, const float* __restrict__ b,
    float* __restrict__ out, int HW, int total)
{
  int idx = blockIdx.x * 256 + threadIdx.x;
  if (idx >= total) return;
  int wcol = idx % HW;
  int t    = idx / HW;
  int hrow = t % HW;
  float s = b[0];
  const _Float16* p = in + (size_t)idx * 64;
#pragma unroll
  for (int c = 0; c < 64; ++c) s += (float)p[c] * w[c];
  out[idx] = (hrow >= wcol) ? s : 0.f;
}

extern "C" void kernel_launch(void* const* d_in, const int* in_sizes, int n_in,
                              void* d_out, int out_size, void* d_ws, size_t ws_size,
                              hipStream_t stream) {
  (void)in_sizes; (void)n_in; (void)out_size; (void)ws_size;
  const float* x  = (const float*)d_in[0];
  const float* w1 = (const float*)d_in[1];
  const float* b1 = (const float*)d_in[2];
  const float* a1 = (const float*)d_in[3];
  const float* w2 = (const float*)d_in[4];
  const float* b2 = (const float*)d_in[5];
  const float* a2 = (const float*)d_in[6];
  const float* w3 = (const float*)d_in[7];
  const float* b3 = (const float*)d_in[8];
  const float* a3 = (const float*)d_in[9];
  const float* w4 = (const float*)d_in[10];
  const float* b4 = (const float*)d_in[11];
  const float* a4 = (const float*)d_in[12];
  const float* w5 = (const float*)d_in[13];
  const float* b5 = (const float*)d_in[14];
  const float* a5 = (const float*)d_in[15];
  const float* w6 = (const float*)d_in[16];
  const float* b6 = (const float*)d_in[17];

  // Workspace carve-up (256B aligned). a4 reuses a2's slot, a5 reuses a1's.
  char* ws = (char*)d_ws;
  size_t off = 0;
  auto carve = [&](size_t bytes) -> void* {
    void* p = ws + off;
    off += (bytes + 255) & ~(size_t)255;
    return p;
  };
  _Float16* act1 = (_Float16*)carve((size_t)4 * 256 * 256 * 64 * 2);   //  33.5 MB
  _Float16* act2 = (_Float16*)carve((size_t)4 * 257 * 255 * 256 * 2);  // 134.2 MB
  _Float16* act3 = (_Float16*)carve((size_t)4 * 258 * 254 * 512 * 2);  // 268.4 MB
  _Float16* w2h  = (_Float16*)carve((size_t)65536 * 2);
  _Float16* w3h  = (_Float16*)carve((size_t)524288 * 2);
  _Float16* w4h  = (_Float16*)carve((size_t)524288 * 2);
  _Float16* w5h  = (_Float16*)carve((size_t)65536 * 2);

  // Weight conversion f32 -> f16 with transpose to co-major [COUT][K].
  cvt_w_t<<<(65536 + 255) / 256, 256, 0, stream>>>(w2, w2h, 256, 256);
  cvt_w_t<<<(524288 + 255) / 256, 256, 0, stream>>>(w3, w3h, 1024, 512);
  cvt_w_t<<<(524288 + 255) / 256, 256, 0, stream>>>(w4, w4h, 2048, 256);
  cvt_w_t<<<(65536 + 255) / 256, 256, 0, stream>>>(w5, w5h, 1024, 64);

  // L1: [4,256,256,1] -> [4,256,256,64]
  {
    int total = 4 * 256 * 256 * 64;
    l1_conv1x1<<<total / 256, 256, 0, stream>>>(x, w1, b1, a1, act1, total);
  }
  // L2: 2x2 pad(1,0): 256x256x64 -> 257x255x256
  {
    int M = 4 * 257 * 255;
    dim3 g((M + 127) / 128, 256 / 64);
    conv2x2_wmma<64, 256, 1, 0><<<g, 256, 0, stream>>>(act1, w2h, b2, a2, act2,
                                                       4, 256, 256);
  }
  // L3: 2x2 pad(1,0): 257x255x256 -> 258x254x512
  {
    int M = 4 * 258 * 254;
    dim3 g((M + 127) / 128, 512 / 64);
    conv2x2_wmma<256, 512, 1, 0><<<g, 256, 0, stream>>>(act2, w3h, b3, a3, act3,
                                                        4, 257, 255);
  }
  // L4: 2x2 pad(0,1): 258x254x512 -> 257x255x256   (writes into act2 slot)
  {
    int M = 4 * 257 * 255;
    dim3 g((M + 127) / 128, 256 / 64);
    conv2x2_wmma<512, 256, 0, 1><<<g, 256, 0, stream>>>(act3, w4h, b4, a4, act2,
                                                        4, 258, 254);
  }
  // L5: 2x2 pad(0,1): 257x255x256 -> 256x256x64    (writes into act1 slot)
  {
    int M = 4 * 256 * 256;
    dim3 g((M + 127) / 128, 64 / 64);
    conv2x2_wmma<256, 64, 0, 1><<<g, 256, 0, stream>>>(act2, w5h, b5, a5, act1,
                                                       4, 257, 255);
  }
  // L6: 1x1 64->1 + strict-upper-triangle zeroing, f32 out
  {
    int total = 4 * 256 * 256;
    l6_conv1x1_mask<<<(total + 255) / 256, 256, 0, stream>>>(
        act1, w6, b6, (float*)d_out, 256, total);
  }
}
